// EnterpriseFFN_26353919328794
// MI455X (gfx1250) — matrix operand order repaired and checked by reading, not compiled, not run on hardware
//
#include <hip/hip_runtime.h>
#include <hip/hip_bf16.h>
#include <math.h>

// ---------------- problem constants ----------------
#define NUM_EXPERTS 8
#define D_MODEL     1024
#define EXPERT_CAP  512          // H
#define NTOK        8192         // B*S = 4*2048
#define M_TILE      64           // tokens per ffn block

typedef __attribute__((ext_vector_type(16))) __bf16 v16bf;
typedef __attribute__((ext_vector_type(8)))  float  v8f;
typedef __attribute__((__vector_size__(16))) int    v4i;

#define AS1 __attribute__((address_space(1)))
#define AS3 __attribute__((address_space(3)))

#if defined(__HIP_DEVICE_COMPILE__) && __has_builtin(__builtin_amdgcn_global_load_async_to_lds_b128)
#define USE_ASYNC 1
#else
#define USE_ASYNC 0
#endif

static __device__ __forceinline__ __bf16 to_bf16(float f) { return (__bf16)f; }

static __device__ __forceinline__ float gelu_exact(float v) {
    return 0.5f * v * (1.0f + erff(v * 0.70710678118654752440f));
}

// copy 16 bytes global -> LDS (async on CDNA5, sync fallback otherwise)
static __device__ __forceinline__ void copy16_g2l(void* lds, const void* g) {
#if USE_ASYNC
    __builtin_amdgcn_global_load_async_to_lds_b128((AS1 v4i*)g, (AS3 v4i*)lds, 0, 0);
#else
    *(int4*)lds = *(const int4*)g;
#endif
}
static __device__ __forceinline__ void async_drain() {
#if USE_ASYNC
    asm volatile("s_wait_asynccnt 0x0" ::: "memory");
#endif
}

// ---------------- kernel 0: zero output + counters ----------------
__global__ void moe_zero_kernel(float* __restrict__ out, int n4, int* __restrict__ cnt) {
    int i = blockIdx.x * blockDim.x + threadIdx.x;
    float4 z = make_float4(0.f, 0.f, 0.f, 0.f);
    for (int idx = i; idx < n4; idx += gridDim.x * blockDim.x)
        ((float4*)out)[idx] = z;
    if (i < NUM_EXPERTS) cnt[i] = 0;
}

// ---------------- conversion kernels: f32 -> bf16 (x row-major, weights pre-swizzled) ----
// xbf: [N, D] row-major bf16
__global__ void moe_conv_x_kernel(const float* __restrict__ x, __bf16* __restrict__ xbf) {
    int i = blockIdx.x * blockDim.x + threadIdx.x;   // one per 8 elements
    const float4* s = (const float4*)x;
    float4 a = s[2 * i], b = s[2 * i + 1];
    union { __bf16 h[8]; int4 v; } u;
    u.h[0] = to_bf16(a.x); u.h[1] = to_bf16(a.y); u.h[2] = to_bf16(a.z); u.h[3] = to_bf16(a.w);
    u.h[4] = to_bf16(b.x); u.h[5] = to_bf16(b.y); u.h[6] = to_bf16(b.z); u.h[7] = to_bf16(b.w);
    ((int4*)xbf)[i] = u.v;
}

// w1f: [E][kc:32][sub:32][lane:32][i:16] bf16  (B-fragment layout per 32-K chunk)
//   lane<16 -> col = sub*16+lane, K = i ; lane>=16 -> col = sub*16+lane-16, K = 16+i
__global__ void moe_conv_w1_kernel(const float* __restrict__ w1, __bf16* __restrict__ w1f) {
    int o = blockIdx.x * blockDim.x + threadIdx.x;   // one per 8 bf16 (i-half)
    int ihalf = o & 1, ln = (o >> 1) & 31, sub = (o >> 6) & 31, kc = (o >> 11) & 31, e = (o >> 16) & 7;
    int k0 = ((ln >= 16) ? 16 : 0) + ihalf * 8;
    const float* src = w1 + ((size_t)e * D_MODEL + kc * 32 + k0) * EXPERT_CAP + sub * 16 + (ln & 15);
    union { __bf16 h[8]; int4 v; } u;
#pragma unroll
    for (int j = 0; j < 8; ++j) u.h[j] = to_bf16(src[(size_t)j * EXPERT_CAP]);
    ((int4*)w1f)[o] = u.v;
}

// w2f: [E][kc:16][sub:64][lane:32][i:16] bf16
__global__ void moe_conv_w2_kernel(const float* __restrict__ w2, __bf16* __restrict__ w2f) {
    int o = blockIdx.x * blockDim.x + threadIdx.x;   // one per 8 bf16
    int ihalf = o & 1, ln = (o >> 1) & 31, sub = (o >> 6) & 63, kc = (o >> 12) & 15, e = (o >> 16) & 7;
    int k0 = ((ln >= 16) ? 16 : 0) + ihalf * 8;
    const float* src = w2 + ((size_t)e * EXPERT_CAP + kc * 32 + k0) * D_MODEL + sub * 16 + (ln & 15);
    union { __bf16 h[8]; int4 v; } u;
#pragma unroll
    for (int j = 0; j < 8; ++j) u.h[j] = to_bf16(src[(size_t)j * D_MODEL]);
    ((int4*)w2f)[o] = u.v;
}

// ---------------- kernel 1: gating (one wave32 per token) ----------------
__global__ void moe_gate_kernel(const float* __restrict__ x,
                                const float* __restrict__ gw,     // [D, E]
                                float* __restrict__ tokw,          // [N]
                                int* __restrict__ cnt,             // [E]
                                int* __restrict__ lists)           // [E, N]
{
    const int wave = (blockIdx.x * blockDim.x + threadIdx.x) >> 5;
    const int lane = threadIdx.x & 31;
    if (wave >= NTOK) return;

    const float* xr = x + (size_t)wave * D_MODEL;
    float acc[NUM_EXPERTS];
#pragma unroll
    for (int e = 0; e < NUM_EXPERTS; ++e) acc[e] = 0.f;

    for (int d = lane; d < D_MODEL; d += 32) {
        float xv = xr[d];
        const float* g = gw + (size_t)d * NUM_EXPERTS;
#pragma unroll
        for (int e = 0; e < NUM_EXPERTS; ++e) acc[e] += xv * g[e];
    }
#pragma unroll
    for (int e = 0; e < NUM_EXPERTS; ++e) {
#pragma unroll
        for (int off = 16; off > 0; off >>= 1)
            acc[e] += __shfl_xor(acc[e], off, 32);
    }
    if (lane == 0) {
        float mx = acc[0];
#pragma unroll
        for (int e = 1; e < NUM_EXPERTS; ++e) mx = fmaxf(mx, acc[e]);
        float p[NUM_EXPERTS], s = 0.f;
#pragma unroll
        for (int e = 0; e < NUM_EXPERTS; ++e) { p[e] = expf(acc[e] - mx); s += p[e]; }
        float inv = 1.f / s;
#pragma unroll
        for (int e = 0; e < NUM_EXPERTS; ++e) p[e] *= inv;
        int i1 = 0;
#pragma unroll
        for (int e = 1; e < NUM_EXPERTS; ++e) if (p[e] > p[i1]) i1 = e;
        int i2 = (i1 == 0) ? 1 : 0;
#pragma unroll
        for (int e = 0; e < NUM_EXPERTS; ++e)
            if (e != i1 && p[e] > p[i2]) i2 = e;

        tokw[wave] = p[i1] + p[i2];
        int pos1 = atomicAdd(&cnt[i1], 1);
        lists[i1 * NTOK + pos1] = wave;
        int pos2 = atomicAdd(&cnt[i2], 1);
        lists[i2 * NTOK + pos2] = wave;
    }
}

// ---------------- kernel 2: fused expert FFN, bf16 WMMA + async LDS staging ----------
struct Stage1 {
    __bf16 A[4][32][16];     // 4 M-subtiles, A-fragment layout
    __bf16 B[32][32][16];    // 32 N-subtiles of W1 K-chunk, B-fragment layout
};
struct Stage2 {
    __bf16 B2[32][32][16];   // 32 N-subtiles (one 512-col half) of W2 K-chunk
};
union SMemU { Stage1 s1; Stage2 s2; };

__global__ __launch_bounds__(256)
void moe_ffn_kernel(const __bf16* __restrict__ xbf,  // [N, D] bf16
                    const __bf16* __restrict__ w1f,  // pre-swizzled bf16 fragments
                    const float*  __restrict__ b1,   // [E, H]
                    const __bf16* __restrict__ w2f,  // pre-swizzled bf16 fragments
                    const float*  __restrict__ b2,   // [E, D]
                    const float*  __restrict__ tokw, // [N]
                    const int*   __restrict__ cnt,   // [E]
                    const int*   __restrict__ lists, // [E, N]
                    float* __restrict__ out)         // [N, D]
{
    __shared__ __align__(16) SMemU smem;
    __shared__ __align__(16) __bf16 hA[16][4][32][16]; // h as GEMM2 A-fragments, 64KB
    __shared__ int   s_tok[M_TILE];
    __shared__ float s_tw[M_TILE];

    const int e    = blockIdx.y;
    const int tile = blockIdx.x;
    const int count = cnt[e];
    if (tile * M_TILE >= count) return;
    int nvalid = count - tile * M_TILE;
    if (nvalid > M_TILE) nvalid = M_TILE;

    const int t = threadIdx.x;
    const int w = t >> 5;       // wave 0..7
    const int lane = t & 31;

    if (t < M_TILE) {
        int idx = tile * M_TILE + t;
        int tok = (idx < count) ? lists[e * NTOK + idx]
                                : lists[e * NTOK + tile * M_TILE]; // pad with a valid token
        s_tok[t] = tok;
        s_tw[t]  = tokw[tok];
    }
    __syncthreads();

    // ---------------- GEMM1: h[64 x 512] = Xg[64 x 1024] @ W1 ----------------
    v8f acc1[4][4] = {};
    for (int kc = 0; kc < D_MODEL / 32; ++kc) {
        const int kbase = kc * 32;
        // A stage: one 16B async copy per thread (gathered bf16 token rows)
        {
            int msub = t >> 6, ln = (t >> 1) & 31, run = t & 1;
            int tok  = s_tok[msub * 16 + (ln & 15)];
            int koff = run * 8 + ((ln >= 16) ? 16 : 0);
            copy16_g2l(&smem.s1.A[msub][ln][run * 8],
                       xbf + (size_t)tok * D_MODEL + kbase + koff);
        }
        // B stage: contiguous 32KB chunk of pre-swizzled W1 fragments
        {
            const __bf16* chunk = w1f + ((size_t)(e * 32 + kc) << 14); // 16384 elems/chunk
            __bf16* dst = &smem.s1.B[0][0][0];
#pragma unroll
            for (int it = 0; it < 8; ++it) {
                int u = it * 256 + t;
                copy16_g2l(dst + u * 8, chunk + u * 8);
            }
        }
        async_drain();
        __syncthreads();

        v16bf a0 = *(const v16bf*)&smem.s1.A[0][lane][0];
        v16bf a1 = *(const v16bf*)&smem.s1.A[1][lane][0];
        v16bf a2 = *(const v16bf*)&smem.s1.A[2][lane][0];
        v16bf a3 = *(const v16bf*)&smem.s1.A[3][lane][0];
#pragma unroll
        for (int ns = 0; ns < 4; ++ns) {
            v16bf bf = *(const v16bf*)&smem.s1.B[w * 4 + ns][lane][0];
            acc1[0][ns] = __builtin_amdgcn_wmma_f32_16x16x32_bf16(
                false, a0, false, bf, (short)0, acc1[0][ns], false, false);
            acc1[1][ns] = __builtin_amdgcn_wmma_f32_16x16x32_bf16(
                false, a1, false, bf, (short)0, acc1[1][ns], false, false);
            acc1[2][ns] = __builtin_amdgcn_wmma_f32_16x16x32_bf16(
                false, a2, false, bf, (short)0, acc1[2][ns], false, false);
            acc1[3][ns] = __builtin_amdgcn_wmma_f32_16x16x32_bf16(
                false, a3, false, bf, (short)0, acc1[3][ns], false, false);
        }
        __syncthreads();
    }

    // ------------- bias + GELU, write h to LDS as GEMM2 A-fragments -------------
#pragma unroll
    for (int ms = 0; ms < 4; ++ms) {
#pragma unroll
        for (int ns = 0; ns < 4; ++ns) {
            v8f a = acc1[ms][ns];
            const int hcol = w * 64 + ns * 16 + (lane & 15);
            const float bias = b1[e * EXPERT_CAP + hcol];
#pragma unroll
            for (int r = 0; r < 8; ++r) {
                int m = ms * 16 + r + ((lane >> 4) << 3);
                float v = gelu_exact(a[r] + bias);
                int kk = hcol & 31, ch = hcol >> 5;
                int ln = (m & 15) + ((kk & 16) ? 16 : 0);
                int ii = kk & 15;
                hA[ch][m >> 4][ln][ii] = to_bf16(v);
            }
        }
    }
    __syncthreads();

    // -------- GEMM2: y[64 x 1024] = h[64 x 512] @ W2, in two 512-col halves --------
    for (int hf = 0; hf < 2; ++hf) {
        v8f acc2[4][4] = {};
        for (int kc = 0; kc < EXPERT_CAP / 32; ++kc) {
            // B stage: contiguous 32KB of pre-swizzled W2 fragments for this half
            {
                const __bf16* chunk = w2f + (((size_t)(e * 16 + kc) * 64 + hf * 32) << 9);
                __bf16* dst = &smem.s2.B2[0][0][0];
#pragma unroll
                for (int it = 0; it < 8; ++it) {
                    int u = it * 256 + t;
                    copy16_g2l(dst + u * 8, chunk + u * 8);
                }
            }
            async_drain();
            __syncthreads();

            v16bf a0 = *(const v16bf*)&hA[kc][0][lane][0];
            v16bf a1 = *(const v16bf*)&hA[kc][1][lane][0];
            v16bf a2 = *(const v16bf*)&hA[kc][2][lane][0];
            v16bf a3 = *(const v16bf*)&hA[kc][3][lane][0];
#pragma unroll
            for (int ns = 0; ns < 4; ++ns) {
                v16bf bf = *(const v16bf*)&smem.s2.B2[w * 4 + ns][lane][0];
                acc2[0][ns] = __builtin_amdgcn_wmma_f32_16x16x32_bf16(
                    false, a0, false, bf, (short)0, acc2[0][ns], false, false);
                acc2[1][ns] = __builtin_amdgcn_wmma_f32_16x16x32_bf16(
                    false, a1, false, bf, (short)0, acc2[1][ns], false, false);
                acc2[2][ns] = __builtin_amdgcn_wmma_f32_16x16x32_bf16(
                    false, a2, false, bf, (short)0, acc2[2][ns], false, false);
                acc2[3][ns] = __builtin_amdgcn_wmma_f32_16x16x32_bf16(
                    false, a3, false, bf, (short)0, acc2[3][ns], false, false);
            }
            __syncthreads();
        }

        // ---------- epilogue for this half: +b2, scale by tok_w, scatter-add ----------
#pragma unroll
        for (int ms = 0; ms < 4; ++ms) {
#pragma unroll
            for (int ns = 0; ns < 4; ++ns) {
                v8f a = acc2[ms][ns];
                const int ncol = hf * 512 + w * 64 + ns * 16 + (lane & 15);
                const float bias = b2[e * D_MODEL + ncol];
#pragma unroll
                for (int r = 0; r < 8; ++r) {
                    int m = ms * 16 + r + ((lane >> 4) << 3);
                    if (m < nvalid) {
                        float v = (a[r] + bias) * s_tw[m];
                        atomicAdd(out + (size_t)s_tok[m] * D_MODEL + ncol, v);
                    }
                }
            }
        }
    }
}

// ---------------- host launcher ----------------
extern "C" void kernel_launch(void* const* d_in, const int* in_sizes, int n_in,
                              void* d_out, int out_size, void* d_ws, size_t ws_size,
                              hipStream_t stream) {
    const float* x      = (const float*)d_in[0];   // [4,2048,1024]
    const float* gate_w = (const float*)d_in[1];   // [1024,8]
    const float* w1     = (const float*)d_in[2];   // [8,1024,512]
    const float* b1     = (const float*)d_in[3];   // [8,512]
    const float* w2     = (const float*)d_in[4];   // [8,512,1024]
    const float* b2     = (const float*)d_in[5];   // [8,1024]
    float* out = (float*)d_out;                    // [8192,1024]

    // workspace layout (~32.3 MB)
    int*   cnt   = (int*)d_ws;                            // [8]
    int*   lists = cnt + NUM_EXPERTS;                     // [8,8192]
    float* tokw  = (float*)(lists + NUM_EXPERTS * NTOK);  // [8192]
    __bf16* xbf  = (__bf16*)(((uintptr_t)(tokw + NTOK) + 255) & ~(uintptr_t)255);
    __bf16* w1f  = xbf + (size_t)NTOK * D_MODEL;                       // 8 MB
    __bf16* w2f  = w1f + (size_t)NUM_EXPERTS * D_MODEL * EXPERT_CAP;   // 8 MB

    // 1) zero output + routing counters
    moe_zero_kernel<<<2048, 256, 0, stream>>>(out, (NTOK * D_MODEL) / 4, cnt);

    // 2) precision conversion + fragment pre-swizzle
    moe_conv_x_kernel<<<(NTOK * D_MODEL / 8) / 256, 256, 0, stream>>>(x, xbf);
    moe_conv_w1_kernel<<<(NUM_EXPERTS * D_MODEL * EXPERT_CAP / 8) / 256, 256, 0, stream>>>(w1, w1f);
    moe_conv_w2_kernel<<<(NUM_EXPERTS * EXPERT_CAP * D_MODEL / 8) / 256, 256, 0, stream>>>(w2, w2f);

    // 3) gating: 8192 wave32s
    moe_gate_kernel<<<NTOK / 8, 256, 0, stream>>>(x, gate_w, tokw, cnt, lists);

    // 4) routed expert FFN: up to 128 tiles of 64 tokens per expert
    dim3 grid(NTOK / M_TILE, NUM_EXPERTS);
    moe_ffn_kernel<<<grid, 256, 0, stream>>>(xbf, w1f, b1, w2f, b2, tokw, cnt, lists, out);
}
